// OldRouterModel_75617194213641
// MI455X (gfx1250) — matrix operand order repaired and checked by reading, not compiled, not run on hardware
//
#include <hip/hip_runtime.h>
#include <hip/hip_bf16.h>

typedef __bf16 bf16_t;
typedef __attribute__((ext_vector_type(16))) __bf16 v16bf;
typedef __attribute__((ext_vector_type(8)))  __bf16 v8bf;
typedef __attribute__((ext_vector_type(8)))  float  v8f;

// Load a 16x32 A-fragment (or 32x16 B-fragment): two contiguous 16-byte chunks.
__device__ __forceinline__ v16bf load_frag(const bf16_t* p0, const bf16_t* p1) {
    union { v16bf v; v8bf h[2]; } u;
    u.h[0] = *(const v8bf*)p0;
    u.h[1] = *(const v8bf*)p1;
    return u.v;
}

__device__ __forceinline__ v8f wmma_bf16(v16bf a, v16bf b, v8f c) {
    return __builtin_amdgcn_wmma_f32_16x16x32_bf16(false, a, false, b, (short)0, c,
                                                   false, false);
}

// ---------------------------------------------------------------- weight prep
// Transpose Wq/Wk/Wv/Wo/Wc1 (128x128) into Wt[n][k] bf16; Wc2 (128x10) -> [16][128]
// (cols 10..15 zero-padded).
__global__ __launch_bounds__(256)
void prep_kernel(const float* __restrict__ Wq, const float* __restrict__ Wk,
                 const float* __restrict__ Wv, const float* __restrict__ Wo,
                 const float* __restrict__ Wc1, const float* __restrict__ Wc2,
                 bf16_t* __restrict__ wt) {
    int mid = blockIdx.x;
    const float* W = (mid == 0) ? Wq : (mid == 1) ? Wk : (mid == 2) ? Wv
                   : (mid == 3) ? Wo : (mid == 4) ? Wc1 : Wc2;
    bf16_t* out = wt + (size_t)mid * 16384;
    if (mid < 5) {
        for (int i = threadIdx.x; i < 16384; i += 256) {
            int n = i >> 7, kk = i & 127;
            out[i] = (bf16_t)W[kk * 128 + n];
        }
    } else {
        for (int i = threadIdx.x; i < 2048; i += 256) {
            int n = i >> 7, kk = i & 127;
            out[i] = (n < 10) ? (bf16_t)W[kk * 10 + n] : (bf16_t)0.f;
        }
    }
}

// ---------------------------------------------------------------- router GEMVs
__global__ __launch_bounds__(256)
void node_score_kernel(const float* __restrict__ nf, const float* __restrict__ Wn,
                       const float* __restrict__ bn, float* __restrict__ ns) {
    int lane = threadIdx.x & 31, wave = threadIdx.x >> 5;
    int row = blockIdx.x * 8 + wave;
    float4 x = *(const float4*)(nf + (size_t)row * 128 + lane * 4);
    float4 w = *(const float4*)(Wn + lane * 4);
    float p = x.x * w.x + x.y * w.y + x.z * w.z + x.w * w.w;
    p += __shfl_xor(p, 16); p += __shfl_xor(p, 8); p += __shfl_xor(p, 4);
    p += __shfl_xor(p, 2);  p += __shfl_xor(p, 1);
    if (lane == 0) ns[row] = p + bn[0];
}

__global__ __launch_bounds__(256)
void edge_score_kernel(const float* __restrict__ efeat, const float* __restrict__ We,
                       const float* __restrict__ be, const float* __restrict__ ns,
                       const int* __restrict__ src, const int* __restrict__ dst,
                       float* __restrict__ es) {
    int lane = threadIdx.x & 31, wave = threadIdx.x >> 5;
    int row = blockIdx.x * 8 + wave;
    float4 x = *(const float4*)(efeat + (size_t)row * 128 + lane * 4);
    float4 w = *(const float4*)(We + lane * 4);
    float p = x.x * w.x + x.y * w.y + x.z * w.z + x.w * w.w;
    p += __shfl_xor(p, 16); p += __shfl_xor(p, 8); p += __shfl_xor(p, 4);
    p += __shfl_xor(p, 2);  p += __shfl_xor(p, 1);
    if (lane == 0)
        es[row] = p + be[0] + 0.5f * (ns[src[row]] + ns[dst[row]]);
}

// ---------------------------------------------------------------- exact top-k
// blockIdx 0: sort 8192 node scores; blockIdx 1: sort 4096 edge scores.
// Ascending bitonic in LDS; k-th largest = sorted[n-k].
__global__ __launch_bounds__(1024)
void topk_kernel(const float* __restrict__ ns, const float* __restrict__ es,
                 float* __restrict__ th) {
    __shared__ float s[8192];
    const int n = (blockIdx.x == 0) ? 8192 : 4096;
    const float* in = (blockIdx.x == 0) ? ns : es;
    const int kidx = (blockIdx.x == 0) ? 4096 : 2048;  // n - k
    for (int i = threadIdx.x; i < n; i += 1024) s[i] = in[i];
    for (int k2 = 2; k2 <= n; k2 <<= 1) {
        for (int j = k2 >> 1; j > 0; j >>= 1) {
            __syncthreads();
            for (int i = threadIdx.x; i < n; i += 1024) {
                int ixj = i ^ j;
                if (ixj > i) {
                    float a = s[i], b = s[ixj];
                    bool up = ((i & k2) == 0);
                    if ((a > b) == up) { s[i] = b; s[ixj] = a; }
                }
            }
        }
    }
    __syncthreads();
    if (threadIdx.x == 0) th[blockIdx.x] = s[kidx];
}

// ---------------------------------------------------------------- ef build
__global__ __launch_bounds__(256)
void ef_kernel(const float* __restrict__ efeat, const float* __restrict__ es,
               const float* __restrict__ ns, const int* __restrict__ src,
               const int* __restrict__ dst, const float* __restrict__ th,
               bf16_t* __restrict__ ef) {
    int idx = blockIdx.x * 256 + threadIdx.x;   // e*128 + d
    int e = idx >> 7;
    float nth = th[0], eth = th[1];
    bool keep = (es[e] >= eth) && (ns[src[e]] >= nth) && (ns[dst[e]] >= nth);
    ef[idx] = keep ? (bf16_t)efeat[idx] : (bf16_t)0.f;
}

// ---------------------------------------------------------------- adjacency bitmask
// adjT[f][ew] bit b : edge f and edge e = ew*32+b share an endpoint.
// One block per key edge f; src/dst staged in LDS once per block.
__global__ __launch_bounds__(128)
void adj_kernel(const int* __restrict__ src, const int* __restrict__ dst,
                unsigned* __restrict__ adjT) {
    __shared__ int ss[4096], sd[4096];
    for (int i = threadIdx.x; i < 4096; i += 128) { ss[i] = src[i]; sd[i] = dst[i]; }
    __syncthreads();
    const int f = blockIdx.x;
    const int ks = ss[f], kd = sd[f];
    unsigned w = 0;
    const int base = threadIdx.x * 32;
#pragma unroll 8
    for (int b = 0; b < 32; ++b) {
        int e = base + b;
        bool a = (ss[e] == ks) | (ss[e] == kd) | (sd[e] == ks) | (sd[e] == kd);
        w |= (unsigned)a << b;
    }
    adjT[(size_t)f * 128 + threadIdx.x] = w;
}

// ---------------------------------------------------------------- WMMA GEMM
// C[M x 16*NT] = A[M x 128](bf16,row-major) * Bt^T + bias, Bt = W^T [128][128] bf16.
// EPI: 0=bf16 store, 1=tanh-gelu+bf16, 2=transposed bf16 (for V^T), 3=f32 first 10 cols.
template <int NT, int EPI>
__global__ __launch_bounds__(256)
void gemm_bf16(const bf16_t* __restrict__ A, const bf16_t* __restrict__ Bt,
               const float* __restrict__ bias, int biasN,
               bf16_t* __restrict__ outb, float* __restrict__ outf, int M) {
    const int lane = threadIdx.x & 31;
    const int wave = threadIdx.x >> 5;
    const int col  = lane & 15;
    const int half = lane >> 4;
    const int e0 = (blockIdx.x * 8 + wave) * 16;
    if (e0 >= M) return;   // wave-uniform

    v16bf a[4];
#pragma unroll
    for (int kc = 0; kc < 4; ++kc) {
        const bf16_t* ap = A + (size_t)(e0 + col) * 128 + 32 * kc;
        a[kc] = load_frag(ap + 8 * half, ap + 16 + 8 * half);
    }
#pragma unroll
    for (int n = 0; n < NT; ++n) {
        int c = n * 16 + col;
        float bv = (c < biasN) ? bias[c] : 0.f;
        v8f acc = {bv, bv, bv, bv, bv, bv, bv, bv};
#pragma unroll
        for (int kc = 0; kc < 4; ++kc) {
            const bf16_t* bp = Bt + (size_t)c * 128 + 32 * kc;
            acc = wmma_bf16(a[kc], load_frag(bp + 8 * half, bp + 16 + 8 * half), acc);
        }
#pragma unroll
        for (int v = 0; v < 8; ++v) {
            int e = e0 + v + 8 * half;
            float x = acc[v];
            if (EPI == 1) {   // jax.nn.gelu (tanh approximation, JAX default)
                float t = tanhf(0.7978845608028654f * (x + 0.044715f * x * x * x));
                x = 0.5f * x * (1.f + t);
            }
            if (EPI == 2)      outb[(size_t)c * M + e] = (bf16_t)x;
            else if (EPI == 3) { if (c < 10) outf[(size_t)e * 10 + c] = x; }
            else               outb[(size_t)e * 128 + c] = (bf16_t)x;
        }
    }
}

// ---------------------------------------------------------------- flash attention
// One wave per (16-query tile, head); 64 keys per iteration.
// Q,K stored [E][128] bf16; V stored transposed [128][E] bf16; adjacency from bitmask.
__global__ __launch_bounds__(128)
void attn_kernel(const bf16_t* __restrict__ q, const bf16_t* __restrict__ k,
                 const bf16_t* __restrict__ vT, const unsigned* __restrict__ adjT,
                 bf16_t* __restrict__ out) {
    const int lane = threadIdx.x & 31;
    const int h    = threadIdx.x >> 5;        // head = wave id
    const int e0   = blockIdx.x * 16;
    const int col  = lane & 15;
    const int half = lane >> 4;

    __shared__ __align__(16) bf16_t pbuf[4][16 * 64];
    bf16_t* pw = pbuf[h];

    const bf16_t* qrow = q + (size_t)(e0 + col) * 128 + h * 32;
    v16bf aq = load_frag(qrow + 8 * half, qrow + 16 + 8 * half);

    float m[8], l[8];
    v8f acc0 = {}, acc1 = {};
#pragma unroll
    for (int v = 0; v < 8; ++v) { m[v] = -__builtin_inff(); l[v] = 0.f; }

    const float scale = 0.17677669529663687f;  // 1/sqrt(32)
    const int ew = e0 >> 5;                    // bitmask word index for query tile
    const int bsh = (e0 & 16) + 8 * half;      // bit offset of this lane's row group

    for (int f0 = 0; f0 < 4096; f0 += 64) {
        float s[4][8];
#pragma unroll
        for (int t = 0; t < 4; ++t) {
            int ft = f0 + 16 * t;
            const bf16_t* kp = k + (size_t)(ft + col) * 128 + h * 32;
            v8f c = {};
            c = wmma_bf16(aq, load_frag(kp + 8 * half, kp + 16 + 8 * half), c);
            unsigned ab = adjT[(size_t)(ft + col) * 128 + ew] >> bsh;
#pragma unroll
            for (int v = 0; v < 8; ++v)
                s[t][v] = ((ab >> v) & 1u) ? c[v] * scale : -1e9f;
        }
#pragma unroll
        for (int v = 0; v < 8; ++v) {
            float rm = fmaxf(fmaxf(s[0][v], s[1][v]), fmaxf(s[2][v], s[3][v]));
            rm = fmaxf(rm, __shfl_xor(rm, 8));
            rm = fmaxf(rm, __shfl_xor(rm, 4));
            rm = fmaxf(rm, __shfl_xor(rm, 2));
            rm = fmaxf(rm, __shfl_xor(rm, 1));
            float nm = fmaxf(m[v], rm);
            float resc = __expf(m[v] - nm);    // m=-inf first iter -> 0
            m[v] = nm;
            float p0 = __expf(s[0][v] - nm);
            float p1 = __expf(s[1][v] - nm);
            float p2 = __expf(s[2][v] - nm);
            float p3 = __expf(s[3][v] - nm);
            float rs = (p0 + p1) + (p2 + p3);
            rs += __shfl_xor(rs, 8);
            rs += __shfl_xor(rs, 4);
            rs += __shfl_xor(rs, 2);
            rs += __shfl_xor(rs, 1);
            l[v] = l[v] * resc + rs;
            acc0[v] *= resc; acc1[v] *= resc;
            int r = (v + 8 * half) * 64 + col;  // C-layout -> LDS (16x64 row-major)
            pw[r]      = (bf16_t)p0;
            pw[r + 16] = (bf16_t)p1;
            pw[r + 32] = (bf16_t)p2;
            pw[r + 48] = (bf16_t)p3;
        }
        __syncthreads();
        const bf16_t* pr = pw + col * 64;
        v16bf ap0 = load_frag(pr + 8 * half,      pr + 16 + 8 * half);   // P cols 0..31
        v16bf ap1 = load_frag(pr + 32 + 8 * half, pr + 48 + 8 * half);   // P cols 32..63
        const bf16_t* v0 = vT + (size_t)(h * 32 + col) * 4096 + f0;
        const bf16_t* v1 = vT + (size_t)(h * 32 + 16 + col) * 4096 + f0;
        acc0 = wmma_bf16(ap0, load_frag(v0 + 8 * half, v0 + 16 + 8 * half), acc0);
        acc0 = wmma_bf16(ap1, load_frag(v0 + 32 + 8 * half, v0 + 48 + 8 * half), acc0);
        acc1 = wmma_bf16(ap0, load_frag(v1 + 8 * half, v1 + 16 + 8 * half), acc1);
        acc1 = wmma_bf16(ap1, load_frag(v1 + 32 + 8 * half, v1 + 48 + 8 * half), acc1);
        __syncthreads();
    }
#pragma unroll
    for (int v = 0; v < 8; ++v) {
        int e = e0 + v + 8 * half;
        float inv = 1.0f / l[v];    // l > 0: every edge is adjacent to itself
        out[(size_t)e * 128 + h * 32 + col]      = (bf16_t)(acc0[v] * inv);
        out[(size_t)e * 128 + h * 32 + 16 + col] = (bf16_t)(acc1[v] * inv);
    }
}

// ---------------------------------------------------------------- launcher
extern "C" void kernel_launch(void* const* d_in, const int* in_sizes, int n_in,
                              void* d_out, int out_size, void* d_ws, size_t ws_size,
                              hipStream_t stream) {
    const float* nf    = (const float*)d_in[0];
    const float* efeat = (const float*)d_in[1];
    const int*   eidx  = (const int*)d_in[2];
    // d_in[3] node_tiers: unused by reference
    const float* Wn  = (const float*)d_in[4];
    const float* bn  = (const float*)d_in[5];
    const float* We  = (const float*)d_in[6];
    const float* be  = (const float*)d_in[7];
    const float* Wq  = (const float*)d_in[8];
    const float* bq  = (const float*)d_in[9];
    const float* Wk  = (const float*)d_in[10];
    const float* bk  = (const float*)d_in[11];
    const float* Wv  = (const float*)d_in[12];
    const float* bv  = (const float*)d_in[13];
    const float* Wo  = (const float*)d_in[14];
    const float* bo  = (const float*)d_in[15];
    const float* Wc1 = (const float*)d_in[16];
    const float* bc1 = (const float*)d_in[17];
    const float* Wc2 = (const float*)d_in[18];
    const float* bc2 = (const float*)d_in[19];

    const int* src = eidx;
    const int* dst = eidx + 4096;

    char* w = (char*)d_ws;
    float*    ns  = (float*)(w + 0);          // 8192 f32
    float*    es  = (float*)(w + 32768);      // 4096 f32
    float*    th  = (float*)(w + 49152);      // 2 f32 (+pad)
    bf16_t*   wt  = (bf16_t*)(w + 50176);     // transposed bf16 weights
    bf16_t*   ef  = (bf16_t*)(w + 218112);    // 4096x128 bf16
    bf16_t*   qm  = (bf16_t*)(w + 1266688);   // 4096x128 bf16
    bf16_t*   km  = (bf16_t*)(w + 2315264);   // 4096x128 bf16
    bf16_t*   vt  = (bf16_t*)(w + 3363840);   // 128x4096 bf16 (V transposed)
    bf16_t*   ao  = (bf16_t*)(w + 4412416);   // 4096x128 bf16 attention out
    bf16_t*   o1  = (bf16_t*)(w + 5460992);   // 4096x128 bf16 after Wo
    bf16_t*   hm  = (bf16_t*)(w + 6509568);   // 4096x128 bf16 after gelu(Wc1)
    unsigned* adj = (unsigned*)(w + 7558144); // 4096x128 u32 adjacency bitmask

    prep_kernel<<<6, 256, 0, stream>>>(Wq, Wk, Wv, Wo, Wc1, Wc2, wt);
    node_score_kernel<<<1024, 256, 0, stream>>>(nf, Wn, bn, ns);
    edge_score_kernel<<<512, 256, 0, stream>>>(efeat, We, be, ns, src, dst, es);
    topk_kernel<<<2, 1024, 0, stream>>>(ns, es, th);
    ef_kernel<<<2048, 256, 0, stream>>>(efeat, es, ns, src, dst, th, ef);
    adj_kernel<<<4096, 128, 0, stream>>>(src, dst, adj);

    gemm_bf16<8, 0><<<32, 256, 0, stream>>>(ef, wt + 0 * 16384, bq, 128, qm, nullptr, 4096);
    gemm_bf16<8, 0><<<32, 256, 0, stream>>>(ef, wt + 1 * 16384, bk, 128, km, nullptr, 4096);
    gemm_bf16<8, 2><<<32, 256, 0, stream>>>(ef, wt + 2 * 16384, bv, 128, vt, nullptr, 4096);

    attn_kernel<<<256, 128, 0, stream>>>(qm, km, vt, adj, ao);

    gemm_bf16<8, 0><<<32, 256, 0, stream>>>(ao, wt + 3 * 16384, bo, 128, o1, nullptr, 4096);
    gemm_bf16<8, 1><<<32, 256, 0, stream>>>(o1, wt + 4 * 16384, bc1, 128, hm, nullptr, 4096);
    gemm_bf16<1, 3><<<32, 256, 0, stream>>>(hm, wt + 5 * 16384, bc2, 10, nullptr,
                                            (float*)d_out, 4096);
    (void)in_sizes; (void)n_in; (void)out_size; (void)ws_size;
}
